// TemporalEncoderShared_61950608277568
// MI455X (gfx1250) — matrix operand order repaired and checked by reading, not compiled, not run on hardware
//
#include <hip/hip_runtime.h>
#include <math.h>

typedef __attribute__((ext_vector_type(16))) _Float16 v16h;
typedef __attribute__((ext_vector_type(8)))  _Float16 v8h;
typedef __attribute__((ext_vector_type(8)))  float    v8f;

#define BB 32    // batch
#define TT 96    // time steps
#define NN 207   // nodes
#define HH 64    // per-direction hidden
#define GG 192   // 3*H gates
#define DD 128   // 2*H

// Bf 73728 + hA 4096 + gh 24576 + gx 8192 + hprev 8192 + biases 1024
#define SMEM1_BYTES 119808

__device__ __forceinline__ float frcp(float x) { return __builtin_amdgcn_rcpf(x); }
__device__ __forceinline__ float sigf(float x) { return frcp(1.0f + __expf(-x)); }
__device__ __forceinline__ float tanhfast(float x) {
  // tanh(x) = 1 - 2/(1+e^{2x}); e->inf => rcp->0 => +1, e->0 => -1 (saturates correctly)
  return 1.0f - 2.0f * frcp(1.0f + __expf(2.0f * x));
}

// ---------------------------------------------------------------------------
// Kernel 1: layer-0 bi-GRU (input size 1). One block per (node, dir).
// Recurrent GEMM [32x64]x[64x192] on v_wmma_f32_16x16x32_f16; Whh^T fragments
// live in VGPRs for the whole 96-step scan; h state in LDS in A-frag order.
// Output out1 is written in WMMA A-fragment order so layer 2 can load its
// input-K fragments directly from global memory:
//   out1F[node][t][mb(2)][kc(4)][lane(32)][16 halves]
// ---------------------------------------------------------------------------
__global__ __launch_bounds__(256) void gru_layer0_kernel(
    const float* __restrict__ x, const float* __restrict__ Wih0,
    const float* __restrict__ Whh0, const float* __restrict__ bih0,
    const float* __restrict__ bhh0, _Float16* __restrict__ out1F) {
  __shared__ v16h  hA[2 * 2 * 32];      // [mb][kc][lane] A-fragments of h (f16)
  __shared__ float gh[BB * GG];         // h @ Whh^T
  __shared__ float hprev[BB * HH];      // f32 h state
  __shared__ float wxb[GG], bxb[GG], bhb[GG];
  __shared__ float xs[BB];

  const int tid  = threadIdx.x;
  const int node = blockIdx.x >> 1;
  const int dir  = blockIdx.x & 1;
  const int lane = tid & 31;
  const int w    = tid >> 5;
  const int mb   = w >> 2;          // M-tile (batch 0-15 / 16-31)
  const int nb0  = 3 * (w & 3);     // first of 3 N-tiles for this wave
  const int wbase = (node * 2 + dir) * GG;

  for (int e = tid; e < GG; e += 256) {
    wxb[e] = Wih0[wbase + e];   // [N,2,3H,1] == [N,2,3H]
    bxb[e] = bih0[wbase + e];
    bhb[e] = bhh0[wbase + e];
  }
  for (int e = tid; e < 2 * 2 * 32; e += 256) { v16h z = {}; hA[e] = z; }
  for (int e = tid; e < BB * HH; e += 256) hprev[e] = 0.0f;

  // B-fragments of Whh^T: B[k][n] = Whh[n][k]; lane&15 -> n, lane>>4 -> K half
  v16h bf[3][2];
  #pragma unroll
  for (int f = 0; f < 3; ++f) {
    int n = 16 * (nb0 + f) + (lane & 15);
    #pragma unroll
    for (int kc = 0; kc < 2; ++kc) {
      int kb = kc * 32 + ((lane >> 4) << 3);
      #pragma unroll
      for (int e = 0; e < 16; ++e)
        bf[f][kc][e] = (_Float16)Whh0[(wbase + n) * HH + kb + e + ((e >> 3) << 3)];
    }
  }
  __syncthreads();

  _Float16* hA_h = (_Float16*)hA;

  // elementwise mapping: each thread owns 8 consecutive K-columns of one row
  const int eb  = tid >> 3;         // batch row 0..31
  const int ejb = (tid & 7) * 8;    // first of 8 consecutive j
  // precompute fragment-slot addresses for vector stores
  const int e_kc    = ejb >> 5;
  const int e_hidxb = (((ejb & 31) >> 4) << 3);
  const int e_lane  = (eb & 15) | ((((ejb & 31) >> 3) & 1) << 4);
  const int e_mb    = eb >> 4;
  _Float16* hA_slot = &hA_h[(((e_mb * 2) + e_kc) * 32 + e_lane) * 16 + e_hidxb];
  const int c0     = dir * HH + ejb;                 // out1 feature base
  const int o_kc   = c0 >> 5;
  const int o_hxb  = (((c0 & 31) >> 4) << 3);
  const int o_lane = (eb & 15) | ((((c0 & 31) >> 3) & 1) << 4);
  const size_t o_off = ((size_t)(e_mb * 4 + o_kc) * 32 + o_lane) * 16 + o_hxb;

  for (int s = 0; s < TT; ++s) {
    const int t = dir ? (TT - 1 - s) : s;
    if (tid < BB) xs[tid] = x[tid * (TT * NN) + t * NN + node];
    __syncthreads();

    // gh = h @ Whh^T
    v16h a0 = hA[(mb * 2 + 0) * 32 + lane];
    v16h a1 = hA[(mb * 2 + 1) * 32 + lane];
    #pragma unroll
    for (int f = 0; f < 3; ++f) {
      v8f acc = {};
      acc = __builtin_amdgcn_wmma_f32_16x16x32_f16(false, a0, false, bf[f][0], (short)0, acc, false, false);
      acc = __builtin_amdgcn_wmma_f32_16x16x32_f16(false, a1, false, bf[f][1], (short)0, acc, false, false);
      int col = 16 * (nb0 + f) + (lane & 15);
      #pragma unroll
      for (int v = 0; v < 8; ++v) {
        int row = 16 * mb + 8 * (lane >> 4) + v;
        gh[row * GG + col] = acc[v];
      }
    }
    __syncthreads();

    // elementwise GRU update (input gates are scalar FMAs, input size 1)
    {
      float xv = xs[eb];
      v8h hv;
      #pragma unroll
      for (int u = 0; u < 8; ++u) {
        int j = ejb + u;
        float r  = sigf(xv * wxb[j] + bxb[j] + gh[eb * GG + j] + bhb[j]);
        float zg = sigf(xv * wxb[64 + j] + bxb[64 + j] + gh[eb * GG + 64 + j] + bhb[64 + j]);
        float hn = gh[eb * GG + 128 + j] + bhb[128 + j];
        float xn = xv * wxb[128 + j] + bxb[128 + j];
        float ng = tanhfast(xn + r * hn);
        float hp = hprev[eb * HH + j];
        float hnew = (1.0f - zg) * ng + zg * hp;
        hprev[eb * HH + j] = hnew;
        hv[u] = (_Float16)hnew;
      }
      *(v8h*)hA_slot = hv;                                    // 16B LDS store
      *(v8h*)(out1F + (size_t)(node * TT + t) * 4096 + o_off) = hv;  // 16B global store
    }
    __syncthreads();
  }
}

// ---------------------------------------------------------------------------
// Kernel 2: layer-1 bi-GRU (input size 2H=128). One block per (node, dir).
// Fused GEMM: A = [h | out1_t] (K=192), B = [Whh1^T ; Wih1^T] (192x192).
// A K-chunks 0..1 (h) come from LDS; chunks 2..5 (out1_t) are loaded directly
// from global as pre-packed 32B fragments (L2 resident). n-gate columns keep
// separate h-part / x-part accumulators (r scales only hn). Forward dir does
// the full 96-step scan; backward needs only 1 step for out2[-1].
// ---------------------------------------------------------------------------
__global__ __launch_bounds__(256) void gru_layer1_kernel(
    const _Float16* __restrict__ out1F, const float* __restrict__ Wih1,
    const float* __restrict__ Whh1, const float* __restrict__ bih1,
    const float* __restrict__ bhh1, float* __restrict__ hlast) {
  extern __shared__ __align__(32) char smem[];
  v16h*  Bf    = (v16h*)smem;                      // 12 nb * 6 kc * 32 = 73728 B
  v16h*  hA    = (v16h*)(smem + 73728);            // 2 mb * 2 kc * 32  =  4096 B
  float* gh    = (float*)(smem + 73728 + 4096);    // 32*192 = 24576 B
  float* gx    = gh + BB * GG;                     // 32*64  =  8192 B (x-part of n gate)
  float* hprev = gx + BB * HH;                     // 32*64  =  8192 B
  float* bsum  = hprev + BB * HH;                  // 128
  float* bhn   = bsum + 128;                       // 64
  float* bxn   = bhn + 64;                         // 64

  const int tid  = threadIdx.x;
  const int node = blockIdx.x >> 1;
  const int dir  = blockIdx.x & 1;
  const int lane = tid & 31;
  const int w    = tid >> 5;
  const int mb   = w >> 2;
  const int nb0  = 3 * (w & 3);
  const int wbase = (node * 2 + dir) * GG;
  const int steps = dir ? 1 : TT;   // out2[-1] backward half = 1 GRU step on out1[T-1]

  // Pack B = [Whh1^T (K 0..63) ; Wih1^T (K 64..191)] into LDS fragments
  for (int idx = tid; idx < 12 * 6 * 32; idx += 256) {
    int lf = idx & 31;
    int kc = (idx >> 5) % 6;
    int nb = idx / (6 * 32);
    int n  = 16 * nb + (lf & 15);
    int kb = kc * 32 + ((lf >> 4) << 3);
    v16h frag;
    #pragma unroll
    for (int e = 0; e < 16; ++e) {
      int k = kb + e + ((e >> 3) << 3);
      float v = (k < HH) ? Whh1[(wbase + n) * HH + k]
                         : Wih1[(wbase + n) * DD + (k - HH)];
      frag[e] = (_Float16)v;
    }
    Bf[(nb * 6 + kc) * 32 + lf] = frag;
  }
  for (int e = tid; e < 128; e += 256) bsum[e] = bih1[wbase + e] + bhh1[wbase + e];
  for (int e = tid; e < 64; e += 256)  { bhn[e] = bhh1[wbase + 128 + e]; bxn[e] = bih1[wbase + 128 + e]; }
  for (int e = tid; e < 2 * 2 * 32; e += 256) { v16h z = {}; hA[e] = z; }
  for (int e = tid; e < BB * HH; e += 256) hprev[e] = 0.0f;
  __syncthreads();

  _Float16* hA_h = (_Float16*)hA;

  // elementwise mapping: 8 consecutive j per thread
  const int eb  = tid >> 3;
  const int ejb = (tid & 7) * 8;
  const int e_kc    = ejb >> 5;
  const int e_hidxb = (((ejb & 31) >> 4) << 3);
  const int e_lane  = (eb & 15) | ((((ejb & 31) >> 3) & 1) << 4);
  const int e_mb    = eb >> 4;
  _Float16* hA_slot = &hA_h[(((e_mb * 2) + e_kc) * 32 + e_lane) * 16 + e_hidxb];

  for (int s = 0; s < steps; ++s) {
    const int t = dir ? (TT - 1 - s) : s;

    v16h a[6];
    a[0] = hA[(mb * 2 + 0) * 32 + lane];
    a[1] = hA[(mb * 2 + 1) * 32 + lane];
    const v16h* aglob = (const v16h*)out1F + ((size_t)(node * TT + t) * 2 + mb) * 4 * 32;
    #pragma unroll
    for (int q = 0; q < 4; ++q) a[2 + q] = aglob[q * 32 + lane];

    #pragma unroll
    for (int f = 0; f < 3; ++f) {
      int nb = nb0 + f;
      if (nb < 8) {
        // r/z gate columns: x and h contributions sum, fuse over full K=192
        v8f acc = {};
        #pragma unroll
        for (int kc = 0; kc < 6; ++kc)
          acc = __builtin_amdgcn_wmma_f32_16x16x32_f16(false, a[kc], false,
                    Bf[(nb * 6 + kc) * 32 + lane], (short)0, acc, false, false);
        int col = 16 * nb + (lane & 15);
        #pragma unroll
        for (int v = 0; v < 8; ++v)
          gh[(16 * mb + 8 * (lane >> 4) + v) * GG + col] = acc[v];
      } else {
        // n-gate columns: keep h-part (hn) and x-part (xn) separate
        v8f accN = {}, accX = {};
        #pragma unroll
        for (int kc = 0; kc < 2; ++kc)
          accN = __builtin_amdgcn_wmma_f32_16x16x32_f16(false, a[kc], false,
                     Bf[(nb * 6 + kc) * 32 + lane], (short)0, accN, false, false);
        #pragma unroll
        for (int kc = 2; kc < 6; ++kc)
          accX = __builtin_amdgcn_wmma_f32_16x16x32_f16(false, a[kc], false,
                     Bf[(nb * 6 + kc) * 32 + lane], (short)0, accX, false, false);
        int coln = lane & 15;
        #pragma unroll
        for (int v = 0; v < 8; ++v) {
          int row = 16 * mb + 8 * (lane >> 4) + v;
          gh[row * GG + 16 * nb + coln]       = accN[v];
          gx[row * HH + 16 * (nb - 8) + coln] = accX[v];
        }
      }
    }
    __syncthreads();

    {
      v8h hv;
      #pragma unroll
      for (int u = 0; u < 8; ++u) {
        int j = ejb + u;
        float r  = sigf(gh[eb * GG + j] + bsum[j]);
        float zg = sigf(gh[eb * GG + 64 + j] + bsum[64 + j]);
        float hn = gh[eb * GG + 128 + j] + bhn[j];
        float xn = gx[eb * HH + j] + bxn[j];
        float ng = tanhfast(xn + r * hn);
        float hp = hprev[eb * HH + j];
        float hnew = (1.0f - zg) * ng + zg * hp;
        hprev[eb * HH + j] = hnew;
        hv[u] = (_Float16)hnew;
      }
      *(v8h*)hA_slot = hv;   // 16B LDS store of 8 consecutive K columns
    }
    __syncthreads();
  }

  for (int e = tid; e < BB * HH; e += 256) {
    int b = e >> 6, j = e & 63;
    hlast[(node * BB + b) * DD + dir * HH + j] = hprev[b * HH + j];
  }
}

// ---------------------------------------------------------------------------
// Kernel 3: gated residual + LayerNorm + ReLU head. ~0.2 GFLOP -> plain VALU.
// ---------------------------------------------------------------------------
__global__ __launch_bounds__(256) void head_kernel(
    const float* __restrict__ x, const float* __restrict__ Wres,
    const float* __restrict__ bres, const float* __restrict__ Wg,
    const float* __restrict__ bg, const float* __restrict__ gamma,
    const float* __restrict__ beta, const float* __restrict__ hlast,
    float* __restrict__ out) {
  __shared__ float hl[BB * DD];
  __shared__ float rv[BB * DD];
  __shared__ float hf[BB * DD];
  __shared__ float mu[BB], rsg[BB];
  const int tid  = threadIdx.x;
  const int node = blockIdx.x;

  for (int e = tid; e < BB * DD; e += 256) {
    int b = e >> 7, d = e & 127;
    hl[e] = hlast[(node * BB + b) * DD + d];
    float xl = x[b * (TT * NN) + (TT - 1) * NN + node];
    rv[e] = xl * Wres[node * DD + d] + bres[node * DD + d];
  }
  __syncthreads();

  for (int e = tid; e < BB * DD; e += 256) {
    int b = e >> 7, d = e & 127;
    const float* wg = Wg + ((size_t)node * DD + d) * (2 * DD);
    float s = bg[node * DD + d];
    for (int c = 0; c < DD; ++c) s += hl[b * DD + c] * wg[c];
    for (int c = 0; c < DD; ++c) s += rv[b * DD + c] * wg[DD + c];
    float g = sigf(s);
    hf[e] = g * hl[e] + (1.0f - g) * rv[e];
  }
  __syncthreads();

  if (tid < BB) {
    float m = 0.0f;
    for (int d = 0; d < DD; ++d) m += hf[tid * DD + d];
    m *= (1.0f / DD);
    float v = 0.0f;
    for (int d = 0; d < DD; ++d) { float q = hf[tid * DD + d] - m; v += q * q; }
    v *= (1.0f / DD);
    mu[tid]  = m;
    rsg[tid] = rsqrtf(v + 1e-5f);
  }
  __syncthreads();

  for (int e = tid; e < BB * DD; e += 256) {
    int b = e >> 7, d = e & 127;
    float hv = (hf[e] - mu[b]) * rsg[b] * gamma[node * DD + d] + beta[node * DD + d];
    out[(b * NN + node) * DD + d] = fmaxf(hv, 0.0f);
  }
}

// ---------------------------------------------------------------------------
extern "C" void kernel_launch(void* const* d_in, const int* in_sizes, int n_in,
                              void* d_out, int out_size, void* d_ws, size_t ws_size,
                              hipStream_t stream) {
  (void)in_sizes; (void)n_in; (void)out_size; (void)ws_size;
  const float* x     = (const float*)d_in[0];
  const float* Wih0  = (const float*)d_in[1];
  const float* Whh0  = (const float*)d_in[2];
  const float* bih0  = (const float*)d_in[3];
  const float* bhh0  = (const float*)d_in[4];
  const float* Wih1  = (const float*)d_in[5];
  const float* Whh1  = (const float*)d_in[6];
  const float* bih1  = (const float*)d_in[7];
  const float* bhh1  = (const float*)d_in[8];
  const float* Wres  = (const float*)d_in[9];
  const float* bres  = (const float*)d_in[10];
  const float* Wg    = (const float*)d_in[11];
  const float* bg    = (const float*)d_in[12];
  const float* gamma = (const float*)d_in[13];
  const float* beta  = (const float*)d_in[14];

  // Workspace: out1 f16 in fragment order [N][T][4096 halves], then hlast f32
  _Float16* out1F = (_Float16*)d_ws;
  size_t out1_bytes = (size_t)NN * TT * 4096 * sizeof(_Float16);
  float* hlast = (float*)((char*)d_ws + out1_bytes);

  hipFuncSetAttribute((const void*)gru_layer1_kernel,
                      hipFuncAttributeMaxDynamicSharedMemorySize, SMEM1_BYTES);

  gru_layer0_kernel<<<dim3(2 * NN), dim3(256), 0, stream>>>(
      x, Wih0, Whh0, bih0, bhh0, out1F);
  gru_layer1_kernel<<<dim3(2 * NN), dim3(256), SMEM1_BYTES, stream>>>(
      out1F, Wih1, Whh1, bih1, bhh1, hlast);
  head_kernel<<<dim3(NN), dim3(256), 0, stream>>>(
      x, Wres, bres, Wg, bg, gamma, beta, hlast, (float*)d_out);
}
